// Downsampleimprove1_18038862643770
// MI455X (gfx1250) — matrix operand order repaired and checked by reading, not compiled, not run on hardware
//
#include <hip/hip_runtime.h>
#include <cstdint>

typedef _Float16 f16;
typedef __attribute__((ext_vector_type(16))) _Float16 v16h;
typedef __attribute__((ext_vector_type(8)))  _Float16 v8h;
typedef __attribute__((ext_vector_type(8)))  float    v8f;
typedef __attribute__((ext_vector_type(4)))  unsigned int u32x4;
typedef __attribute__((ext_vector_type(8)))  int i32x8;
typedef __attribute__((ext_vector_type(4)))  int i32x4;

namespace {
constexpr int Nn = 4, Cc = 2048, Oo = 256, Ll = 9;
constexpr int HW = 1024;           // 32x32
constexpr int KT = Ll * Cc;        // 18432
constexpr int WP = 34;             // padded image row width (guard col 0 and 33)
constexpr float BN_EPS = 1e-5f;
}

// ---- pre-pass 1: x (N,C,H,W) f32 -> x_p (N*H, 34, C) f16, channels-last, x-padded ----
__global__ void k_transpose(const float* __restrict__ x, f16* __restrict__ xp) {
  __shared__ float tile[32][33];
  const int bc = blockIdx.x, bp = blockIdx.y;
  const int tx = threadIdx.x & 31, ty = threadIdx.x >> 5;
#pragma unroll
  for (int i = 0; i < 4; ++i) {
    int c = bc * 32 + ty + i * 8;
    int p = bp * 32 + tx;                 // p = n*1024 + hw
    int n = p >> 10, hw = p & 1023;
    tile[ty + i * 8][tx] = x[n * (Cc * HW) + c * HW + hw];
  }
  __syncthreads();
#pragma unroll
  for (int i = 0; i < 4; ++i) {
    int p = bp * 32 + ty + i * 8;
    int c = bc * 32 + tx;
    int n = p >> 10, yy = (p >> 5) & 31, xx = p & 31;
    xp[((n * 32 + yy) * WP + xx + 1) * Cc + c] = (f16)tile[tx][ty + i * 8];
  }
}

// ---- pre-pass 1b: zero the guard columns (xp==0 and xp==33) ----
__global__ void k_zero_guard(f16* __restrict__ xp) {
  int t = blockIdx.x * 256 + threadIdx.x;       // (n*32+y)*2*Cc + side*Cc + c
  int row = t / (2 * Cc), rem = t % (2 * Cc);
  int side = rem / Cc, c = rem % Cc;
  xp[(row * WP + (side ? 33 : 0)) * Cc + c] = (f16)0.f;
}

// ---- pre-pass 2: weight (O,C,9) f32 -> Bp[o][k] f16, k = l*C + c ----
__global__ void k_pack_w(const float* __restrict__ w, f16* __restrict__ Bp) {
  int t = blockIdx.x * 256 + threadIdx.x;
  int o = t / KT, rem = t % KT;
  int l = rem / Cc, c = rem % Cc;
  Bp[t] = (f16)w[(o * Cc + c) * Ll + l];
}

// ---- pre-pass 3: conv_w (9,C,3,3) f32 -> Bc[j][k] f16 (j<16, zero-padded) ----
__global__ void k_pack_cw(const float* __restrict__ cw, f16* __restrict__ Bc) {
  int t = blockIdx.x * 256 + threadIdx.x;
  int j = t / KT, rem = t % KT;
  int tap = rem / Cc, c = rem % Cc;
  Bc[t] = (j < 9) ? (f16)cw[(j * Cc + c) * Ll + tap] : (f16)0.f;
}

// ---- TDM: load one 32x32-channel A tile (row stride 2048 elems) into LDS,
//      padded 16 data DWORDs + 4 pad DWORDs -> 80B LDS row stride (bank-conflict free).
__device__ __forceinline__ void tdm_issue(const f16* gsrc, unsigned lds_off) {
  u32x4 g0;
  unsigned long long ga = (unsigned long long)(uintptr_t)gsrc;
  g0[0] = 1u;                                            // count=1
  g0[1] = lds_off;                                       // lds_addr (bytes)
  g0[2] = (unsigned)(ga & 0xffffffffu);                  // global_addr[31:0]
  g0[3] = (unsigned)((ga >> 32) & 0x01ffffffu) | (2u << 30);  // addr[56:32] | type=2
  // data_size=2B, pad_enable, pad_interval=16 DW, pad_amount=4 DW,
  // tensor_dim0=32, tensor_dim1=32, tile_dim0=32, tile_dim1=32, stride0=2048
  i32x8 g1 = {0x06D10000, 0x00200000, 0x00200000, 0x00200000, 0x00000020, 2048, 0, 0};
  i32x4 gz4 = {0, 0, 0, 0};
  i32x8 gz8 = {0, 0, 0, 0, 0, 0, 0, 0};
  __builtin_amdgcn_tensor_load_to_lds(g0, g1, gz4, gz4, gz8, 0);
}

__device__ __forceinline__ bool l_valid(int l, int y) {
  int yy = y + l / 3 - 1;
  return (unsigned)yy < 32u;
}
__device__ __forceinline__ int next_l(int l, int y) {
  do { ++l; } while (l < 9 && !l_valid(l, y));
  return l;
}
__device__ __forceinline__ const f16* tile_src(const f16* xp, int n, int y, int l, int c0) {
  int yy = y + l / 3 - 1;
  return xp + ((n * 32 + yy) * WP + (l % 3)) * Cc + c0;   // xp col = dx+1 = l%3
}

// ---- conv generator as WMMA GEMM: sigma[r, j] = patches[r, k] * Bc[k, j] ----
__global__ __launch_bounds__(256) void k_conv_gemm(
    const f16* __restrict__ xp, const f16* __restrict__ Bc, float* __restrict__ sig)
{
  __shared__ __align__(16) f16 At[2][32][40];
  const int tid = threadIdx.x, lane = tid & 31, wv = tid >> 5;
  const int rb = blockIdx.x, n = rb >> 5, y = rb & 31;
  const int Mr = (wv & 1) * 16 + (lane & 15);
  const int k0h = (lane < 16) ? 0 : 16;      // A: lane-half -> K 0..15 / 16..31
  const int col = lane & 15;                 // B: lane -> N
  const int klo = (lane >> 4) << 4;          // B: lane-half -> K 0..15 / 16..31
  v8f acc = {};
  int l = 0;
  while (l < 9 && !l_valid(l, y)) ++l;
  int c0 = 0, buf = 0;
  if (wv == 0 && l < 9)
    tdm_issue(tile_src(xp, n, y, l, c0), (unsigned)(uintptr_t)&At[0][0][0]);
  while (l < 9) {
    int ln = l, cn = c0 + 32;
    if (cn == Cc) { cn = 0; ln = next_l(l, y); }
    if (wv == 0) __builtin_amdgcn_s_wait_tensorcnt(0);
    __syncthreads();
    if (wv == 0 && ln < 9)
      tdm_issue(tile_src(xp, n, y, ln, cn), (unsigned)(uintptr_t)&At[buf ^ 1][0][0]);
    if (wv < 2) {                            // wave-uniform; EXEC all-1 inside
      v8h alo = *(const v8h*)&At[buf][Mr][k0h];
      v8h ahi = *(const v8h*)&At[buf][Mr][k0h + 8];
      v16h a;
#pragma unroll
      for (int i = 0; i < 8; ++i) { a[i] = alo[i]; a[i + 8] = ahi[i]; }
      v16h b = *(const v16h*)(Bc + col * KT + l * Cc + c0 + klo);
      acc = __builtin_amdgcn_wmma_f32_16x16x32_f16(false, a, false, b,
                                                   (short)0, acc, false, false);
    }
    l = ln; c0 = cn; buf ^= 1;
  }
  if (wv < 2 && col < 9) {
    const int mbase = y * 32 + (wv & 1) * 16 + ((lane >> 4) << 3);
#pragma unroll
    for (int i = 0; i < 8; ++i)
      sig[(n * Ll + col) * HW + mbase + i] = acc[i];
  }
}

// ---- BN statistics: mean / rstd per l over (n,h,w) ----
__global__ void k_stats(const float* __restrict__ sig, float* __restrict__ stats) {
  __shared__ float s1[256], s2[256];
  const int l = blockIdx.x, t = threadIdx.x;
  float a = 0.f, b = 0.f;
  for (int n = 0; n < Nn; ++n)
    for (int m = t; m < HW; m += 256) {
      float v = sig[(n * Ll + l) * HW + m];
      a += v; b += v * v;
    }
  s1[t] = a; s2[t] = b; __syncthreads();
  for (int off = 128; off; off >>= 1) {
    if (t < off) { s1[t] += s1[t + off]; s2[t] += s2[t + off]; }
    __syncthreads();
  }
  if (t == 0) {
    float mean = s1[0] * (1.f / (Nn * HW));
    float var  = s2[0] * (1.f / (Nn * HW)) - mean * mean;
    stats[l * 2]     = mean;
    stats[l * 2 + 1] = rsqrtf(var + BN_EPS);
  }
}

// ---- BN affine + softmax over the 9 taps, output f16 per (n,l,m) ----
__global__ void k_softmax(const float* __restrict__ sig, const float* __restrict__ stats,
                          const float* __restrict__ gamma, const float* __restrict__ beta,
                          f16* __restrict__ ssoft) {
  const int t = blockIdx.x * 256 + threadIdx.x;
  const int n = t >> 10, m = t & 1023;
  float v[9], mx = -1e30f;
#pragma unroll
  for (int l = 0; l < 9; ++l) {
    float s = sig[(n * Ll + l) * HW + m];
    s = (s - stats[l * 2]) * stats[l * 2 + 1] * gamma[l] + beta[l];
    v[l] = s; mx = fmaxf(mx, s);
  }
  float sum = 0.f;
#pragma unroll
  for (int l = 0; l < 9; ++l) { v[l] = __expf(v[l] - mx); sum += v[l]; }
  const float inv = 1.f / sum;
#pragma unroll
  for (int l = 0; l < 9; ++l) ssoft[(n * Ll + l) * HW + m] = (f16)(v[l] * inv);
}

// ---- main GEMM: per-l partial WMMA accumulation, softmax scale folded into the
//      accumulator (per-row scalar) so the A tile is a pure TDM copy.
__global__ __launch_bounds__(256) void k_main_gemm(
    const f16* __restrict__ xp, const f16* __restrict__ Bp,
    const f16* __restrict__ ssoft, float* __restrict__ out)
{
  __shared__ __align__(16) f16 At[2][32][40];
  const int tid = threadIdx.x, lane = tid & 31, wv = tid >> 5;
  const int rb = blockIdx.x, n = rb >> 5, y = rb & 31;
  const int mhalf = wv >> 2, cg = wv & 3;
  const int Mr = mhalf * 16 + (lane & 15);
  const int k0h = (lane < 16) ? 0 : 16;
  const int obc = cg * 64 + (lane & 15);
  const int klo = (lane >> 4) << 4;
  const int mbase = y * 32 + mhalf * 16 + ((lane >> 4) << 3);

  v8f acc[4] = {};
  v8f part[4] = {};

  int l = 0;
  while (l < 9 && !l_valid(l, y)) ++l;
  int c0 = 0, buf = 0;
  if (wv == 0 && l < 9)
    tdm_issue(tile_src(xp, n, y, l, c0), (unsigned)(uintptr_t)&At[0][0][0]);
  while (l < 9) {
    int ln = l, cn = c0 + 32;
    if (cn == Cc) { cn = 0; ln = next_l(l, y); }
    if (wv == 0) __builtin_amdgcn_s_wait_tensorcnt(0);
    __syncthreads();
    if (wv == 0 && ln < 9)
      tdm_issue(tile_src(xp, n, y, ln, cn), (unsigned)(uintptr_t)&At[buf ^ 1][0][0]);

    v8h alo = *(const v8h*)&At[buf][Mr][k0h];
    v8h ahi = *(const v8h*)&At[buf][Mr][k0h + 8];
    v16h a;
#pragma unroll
    for (int i = 0; i < 8; ++i) { a[i] = alo[i]; a[i + 8] = ahi[i]; }
    const f16* b = Bp + obc * KT + l * Cc + c0 + klo;
    v16h b0 = *(const v16h*)(b);
    v16h b1 = *(const v16h*)(b + 16 * KT);
    v16h b2 = *(const v16h*)(b + 32 * KT);
    v16h b3 = *(const v16h*)(b + 48 * KT);
    part[0] = __builtin_amdgcn_wmma_f32_16x16x32_f16(false, a, false, b0, (short)0, part[0], false, false);
    part[1] = __builtin_amdgcn_wmma_f32_16x16x32_f16(false, a, false, b1, (short)0, part[1], false, false);
    part[2] = __builtin_amdgcn_wmma_f32_16x16x32_f16(false, a, false, b2, (short)0, part[2], false, false);
    part[3] = __builtin_amdgcn_wmma_f32_16x16x32_f16(false, a, false, b3, (short)0, part[3], false, false);

    if (c0 == Cc - 32) {   // last K-chunk of this tap: fold softmax scale per row
      v8h s8 = *(const v8h*)&ssoft[(n * Ll + l) * HW + mbase];
#pragma unroll
      for (int j = 0; j < 4; ++j) {
#pragma unroll
        for (int i = 0; i < 8; ++i) acc[j][i] += (float)s8[i] * part[j][i];
        part[j] = (v8f){};
      }
    }
    l = ln; c0 = cn; buf ^= 1;
  }
  const int o0 = cg * 64 + (lane & 15);
#pragma unroll
  for (int j = 0; j < 4; ++j) {
    float* op = out + (n * Oo + o0 + j * 16) * HW + mbase;
#pragma unroll
    for (int i = 0; i < 8; ++i) op[i] = acc[j][i];
  }
}

extern "C" void kernel_launch(void* const* d_in, const int* in_sizes, int n_in,
                              void* d_out, int out_size, void* d_ws, size_t ws_size,
                              hipStream_t stream) {
  (void)in_sizes; (void)n_in; (void)out_size; (void)ws_size;
  const float* x     = (const float*)d_in[0];   // (4,2048,32,32)
  const float* cw    = (const float*)d_in[1];   // (9,2048,3,3)
  const float* gamma = (const float*)d_in[2];   // (9)
  const float* beta  = (const float*)d_in[3];   // (9)
  const float* wgt   = (const float*)d_in[4];   // (256,2048,9)
  float* out = (float*)d_out;                   // (4,256,32,32)

  char* ws = (char*)d_ws;
  f16*   xp    = (f16*)(ws);                    // 4*32*34*2048*2 = 17,825,792 B
  f16*   Bp    = (f16*)(ws + 17825792);         //  9,437,184 B
  f16*   Bc    = (f16*)(ws + 27262976);         //    589,824 B
  float* sig   = (float*)(ws + 27852800);       //    147,456 B
  float* stats = (float*)(ws + 28000256);       //        256 B
  f16*   ssoft = (f16*)(ws + 28000512);         //     73,728 B

  k_transpose <<<dim3(Cc / 32, Nn * HW / 32), 256, 0, stream>>>(x, xp);
  k_zero_guard<<<(Nn * 32 * 2 * Cc) / 256, 256, 0, stream>>>(xp);
  k_pack_w    <<<(Oo * KT) / 256, 256, 0, stream>>>(wgt, Bp);
  k_pack_cw   <<<(16 * KT) / 256, 256, 0, stream>>>(cw, Bc);
  k_conv_gemm <<<Nn * 32, 256, 0, stream>>>(xp, Bc, sig);
  k_stats     <<<Ll, 256, 0, stream>>>(sig, stats);
  k_softmax   <<<(Nn * HW) / 256, 256, 0, stream>>>(sig, stats, gamma, beta, ssoft);
  k_main_gemm <<<Nn * 32, 256, 0, stream>>>(xp, Bp, ssoft, out);
}
